// DistributedAFNO2D_15221364097800
// MI455X (gfx1250) — matrix operand order repaired and checked by reading, not compiled, not run on hardware
//
#include <hip/hip_runtime.h>
#include <hip/hip_bf16.h>
#include <math.h>

// ---------------- problem constants ----------------
#define NB     8
#define BS     96
#define C_DIM  768
#define H_DIM  256
#define W_DIM  512
#define WC     257      // W/2+1
#define KEPT   129      // kept W modes
#define LAM_C  0.01f

typedef __attribute__((ext_vector_type(16))) _Float16 v16h;
typedef __attribute__((ext_vector_type(8)))  _Float16 v8h;
typedef __attribute__((ext_vector_type(8)))  float    v8f;

struct __align__(8) f2 { float x, y; };
__device__ __forceinline__ f2 mkf2(float a, float b) { f2 r; r.x = a; r.y = b; return r; }

// ---------------- batched Stockham radix-2 FFT in LDS ----------------
template<int N>
__device__ __forceinline__ f2* stockham_fft(f2* __restrict__ bufA, f2* __restrict__ bufB,
                                            int batch, float sign, int tid, int nthreads) {
  f2* src = bufA;
  f2* dst = bufB;
  const int half = N / 2;
  for (int Ns = 1; Ns < N; Ns *= 2) {
    __syncthreads();
    int total = half * batch;
    for (int idx = tid; idx < total; idx += nthreads) {
      int b  = idx / half;
      int j  = idx - b * half;
      int jm = j & (Ns - 1);
      float ang = sign * (6.28318530717958647692f * (float)jm / (float)(2 * Ns));
      float sn, cs;
      __sincosf(ang, &sn, &cs);
      f2 u = src[b * N + j];
      f2 v = src[b * N + j + half];
      float tr = cs * v.x - sn * v.y;
      float ti = cs * v.y + sn * v.x;
      int d = ((j - jm) << 1) + jm;           // (j/Ns)*2Ns + (j%Ns)
      dst[b * N + d]      = mkf2(u.x + tr, u.y + ti);
      dst[b * N + d + Ns] = mkf2(u.x - tr, u.y - ti);
    }
    f2* t = src; src = dst; dst = t;
  }
  __syncthreads();
  return src;
}

// ---------------- K0: expand complex weights to f16, fragment-swizzled ----------------
// Logical real matrix (K=192 x N=192):
//   B[i][o]=Re w, B[96+i][o]=-Im w, B[i][96+o]=Im w, B[96+i][96+o]=Re w
// Stored fragment-major for wave32 WMMA B operands:
//   half index = ((((kk*12 + nt)*2 + jh)*32) + lane)*8 + jj
//   kh = (lane>=16)?8:0, i2 = kk*32+kh+jh*16+jj, o2 = nt*16+(lane&15)
__global__ void __launch_bounds__(256)
afno_expand_kernel(const float* __restrict__ w1, const float* __restrict__ b1,
                   const float* __restrict__ w2, const float* __restrict__ b2,
                   _Float16* __restrict__ wexp, float* __restrict__ bexp) {
  const int k = blockIdx.x;
  const int L = blockIdx.y;
  const float* w = L ? w2 : w1;
  const float* b = L ? b2 : b1;
  _Float16* W = wexp + (size_t)(k * 2 + L) * (192 * 192);
  float* Bb = bexp + (k * 2 + L) * 192;
  for (int idx = threadIdx.x; idx < 192 * 192; idx += 256) {
    int jj   = idx & 7;
    int t    = idx >> 3;
    int lane = t & 31;
    t >>= 5;
    int jh = t & 1;
    t >>= 1;
    int nt = t % 12;
    int kk = t / 12;
    int j  = jh * 8 + jj;
    int kh = (lane >= 16) ? 8 : 0;
    int i2 = kk * 32 + kh + jh * 16 + jj;
    int o2 = nt * 16 + (lane & 15);
    int i = i2 - (i2 >= 96 ? 96 : 0);
    int o = o2 - (o2 >= 96 ? 96 : 0);
    const float* pw = w + ((size_t)(k * 96 + i) * 96 + o) * 2;
    float re = pw[0], im = pw[1];
    float v = (i2 < 96) ? ((o2 < 96) ? re : im) : ((o2 < 96) ? -im : re);
    W[idx] = (_Float16)v;
  }
  if (threadIdx.x < 192) {
    int o2 = threadIdx.x;
    int o = o2 - (o2 >= 96 ? 96 : 0);
    Bb[o2] = b[((size_t)k * 96 + o) * 2 + (o2 < 96 ? 0 : 1)];
  }
}

// ---------------- K1: row rfft (512 real -> 257 complex), ortho 1/sqrt(512) ----------------
__global__ void __launch_bounds__(256)
afno_rowfft_kernel(const float* __restrict__ x, f2* __restrict__ spec) {
  extern __shared__ char sm[];
  f2* bufA = (f2*)sm;
  f2* bufB = bufA + W_DIM;
  const int row = blockIdx.x;            // c*256 + h
  const int tid = threadIdx.x;
  const float* xr = x + (size_t)row * W_DIM;
  bufA[tid]       = mkf2(xr[tid], 0.0f);
  bufA[tid + 256] = mkf2(xr[tid + 256], 0.0f);
  f2* res = stockham_fft<W_DIM>(bufA, bufB, 1, -1.0f, tid, 256);
  const float s = 0.04419417382415922f;  // 1/sqrt(512)
  f2* o = spec + (size_t)row * WC;
  f2 v = res[tid];
  o[tid] = mkf2(v.x * s, v.y * s);
  if (tid == 0) {
    f2 w = res[256];
    o[256] = mkf2(w.x * s, w.y * s);
  }
}

// ---------------- K2/K4: column FFT over H for kept columns, in place ----------------
__global__ void __launch_bounds__(256)
afno_colfft_kernel(f2* __restrict__ spec, float sign) {
  extern __shared__ char sm[];
  f2* bufA = (f2*)sm;                    // 16 * 256
  f2* bufB = bufA + 16 * H_DIM;
  const int c  = blockIdx.x;
  const int w0 = blockIdx.y * 16;
  const int ncols = (KEPT - w0) < 16 ? (KEPT - w0) : 16;
  const int tid = threadIdx.x;
  for (int idx = tid; idx < H_DIM * ncols; idx += 256) {
    int h = idx / ncols, cc = idx - h * ncols;
    bufA[cc * H_DIM + h] = spec[((size_t)c * H_DIM + h) * WC + w0 + cc];
  }
  f2* res = stockham_fft<H_DIM>(bufA, bufB, ncols, sign, tid, 256);
  const float s = 0.0625f;               // 1/sqrt(256)
  for (int idx = tid; idx < H_DIM * ncols; idx += 256) {
    int h = idx / ncols, cc = idx - h * ncols;
    f2 v = res[cc * H_DIM + h];
    spec[((size_t)c * H_DIM + h) * WC + w0 + cc] = mkf2(v.x * s, v.y * s);
  }
}

// ---------------- K3: block-diagonal complex MLP via f16 WMMA ----------------
#define AROW 200            // f16 row stride for A tiles (400B, 16B-aligned)
#define OFF_A1   0u         // 64 x AROW f16  = 25600B
#define OFF_A2   25600u     // 64 x AROW f16  = 25600B
#define OFF_BW   51200u     // 192x192 f16 swizzled = 73728B
#define OFF_BIAS 124928u    // 2 x 192 f32 = 1536B
#define MLP_LDS  126464u

// A fragment, documented 16-bit 16x32 layout:
// lanes 0-15: row M=lane, halves = K0..7,K16..23 ; lanes 16-31: K8..15,K24..31
__device__ __forceinline__ v16h load_a_frag(const _Float16* __restrict__ A,
                                            int row0, int k0, int lm, int khalf) {
  const _Float16* p = A + (row0 + lm) * AROW + k0 + khalf;   // khalf: 0 or 8
  v8h lo = *(const v8h*)p;
  v8h hi = *(const v8h*)(p + 16);
  v16h r;
#pragma unroll
  for (int i = 0; i < 8; ++i) { r[i] = lo[i]; r[8 + i] = hi[i]; }
  return r;
}

// B fragment from swizzled store: two conflict-free ds_load_b128 per lane.
__device__ __forceinline__ v16h load_b_frag_swz(const _Float16* __restrict__ Bw,
                                                int frag, int lane) {
  const _Float16* p = Bw + ((size_t)(frag * 2) * 32 + lane) * 8;
  v8h lo = *(const v8h*)p;
  v8h hi = *(const v8h*)(p + 32 * 8);
  v16h r;
#pragma unroll
  for (int i = 0; i < 8; ++i) { r[i] = lo[i]; r[8 + i] = hi[i]; }
  return r;
}

// CDNA5 async global->LDS b128 (ASYNCcnt path). lds_off is a raw LDS byte
// offset (dynamic LDS starts at 0: no static __shared__ in this kernel).
__device__ __forceinline__ void async_g2l_b128(unsigned lds_off, const void* sbase,
                                               unsigned gbyte_off) {
  asm volatile("global_load_async_to_lds_b128 %0, %1, %2"
               :
               : "v"(lds_off), "v"(gbyte_off), "s"(sbase)
               : "memory");
}
__device__ __forceinline__ void wait_asynccnt0() {
  asm volatile("s_wait_asynccnt 0x0" ::: "memory");
}

__global__ void __launch_bounds__(256)
afno_mlp_kernel(f2* __restrict__ spec, const _Float16* __restrict__ wexp,
                const float* __restrict__ bexp) {
  extern __shared__ char sm[];
  _Float16* A1 = (_Float16*)(sm + OFF_A1);
  _Float16* A2 = (_Float16*)(sm + OFF_A2);
  _Float16* Bw = (_Float16*)(sm + OFF_BW);
  float* bias  = (float*)(sm + OFF_BIAS);
  float* stag  = (float*)sm;             // 64 x 200 f32 epilogue staging (overlays A1+A2)

  const int bid = blockIdx.x;
  const int k   = bid / 516;             // 33024/64 = 516 position tiles per block
  const int pt  = bid - k * 516;
  const int p0  = pt * 64;
  const int c0  = k * BS;
  const int tid = threadIdx.x;

  const _Float16* Wg1 = wexp + (size_t)(k * 2 + 0) * (192 * 192);
  const _Float16* Wg2 = wexp + (size_t)(k * 2 + 1) * (192 * 192);
  __builtin_prefetch(Wg2 + tid * 144, 0, 1);   // global_prefetch_b8: warm layer-2 weights

  if (tid < 192) {
    bias[tid]       = bexp[(k * 2 + 0) * 192 + tid];
    bias[192 + tid] = bexp[(k * 2 + 1) * 192 + tid];
  }
  // async-stage layer-1 weights into LDS: 73728B = 4608 b128, 18 per thread
#pragma unroll
  for (int i = 0; i < 18; ++i) {
    unsigned e = (unsigned)(tid + i * 256) * 16u;
    async_g2l_b128(OFF_BW + e, (const void*)Wg1, e);
  }
  // A1: rows = positions, cols = [Re a_0..95 | Im a_0..95]
  for (int idx = tid; idx < 64 * BS; idx += 256) {
    int i = idx >> 6;
    int p = idx & 63;
    int P = p0 + p;
    int eta = P / KEPT;
    int w = P - eta * KEPT;
    f2 v = spec[((size_t)(c0 + i) * H_DIM + eta) * WC + w];
    A1[p * AROW + i]      = (_Float16)v.x;
    A1[p * AROW + BS + i] = (_Float16)v.y;
  }
  wait_asynccnt0();
  __syncthreads();

  const int wid    = tid >> 5;           // wave32: 8 waves
  const int lane   = tid & 31;
  const int lm     = lane & 15;
  const int khalfA = (lane >= 16) ? 8 : 0;
  const int rtile  = (wid & 3) * 16;     // position tile
  const int obase  = (wid >> 2) * 6;     // 6 output tiles per wave
  const int mbase  = rtile + ((lane >= 16) ? 8 : 0);
  const v8f vzero = {};

  // ---- layer 1: t1 = A1 * W1 + b1, ReLU -> A2 (f16) ----
  v8f acc1[6];
#pragma unroll
  for (int ot = 0; ot < 6; ++ot) acc1[ot] = vzero;
#pragma unroll
  for (int kk = 0; kk < 6; ++kk) {
    v16h af = load_a_frag(A1, rtile, kk * 32, lm, khalfA);
#pragma unroll
    for (int ot = 0; ot < 6; ++ot) {
      v16h bf = load_b_frag_swz(Bw, kk * 12 + obase + ot, lane);
      acc1[ot] = __builtin_amdgcn_wmma_f32_16x16x32_f16(false, af, false, bf,
                                                        (short)0, acc1[ot], false, false);
    }
  }
#pragma unroll
  for (int ot = 0; ot < 6; ++ot) {
    int col = (obase + ot) * 16 + lm;
    float bv = bias[col];
#pragma unroll
    for (int r = 0; r < 8; ++r) {
      float v = acc1[ot][r] + bv;
      v = v > 0.0f ? v : 0.0f;
      A2[(mbase + r) * AROW + col] = (_Float16)v;
    }
  }
  __syncthreads();
  // async-stage layer-2 weights (overwrite Bw after barrier)
#pragma unroll
  for (int i = 0; i < 18; ++i) {
    unsigned e = (unsigned)(tid + i * 256) * 16u;
    async_g2l_b128(OFF_BW + e, (const void*)Wg2, e);
  }
  wait_asynccnt0();
  __syncthreads();

  // ---- layer 2: keep accumulators in registers until all A2 reads complete ----
  v8f acc2[6];
#pragma unroll
  for (int ot = 0; ot < 6; ++ot) acc2[ot] = vzero;
#pragma unroll
  for (int kk = 0; kk < 6; ++kk) {
    v16h af = load_a_frag(A2, rtile, kk * 32, lm, khalfA);
#pragma unroll
    for (int ot = 0; ot < 6; ++ot) {
      v16h bf = load_b_frag_swz(Bw, kk * 12 + obase + ot, lane);
      acc2[ot] = __builtin_amdgcn_wmma_f32_16x16x32_f16(false, af, false, bf,
                                                        (short)0, acc2[ot], false, false);
    }
  }
  __syncthreads();                       // A2/Bw dead; staging overlay now safe

#pragma unroll
  for (int ot = 0; ot < 6; ++ot) {
    int col = (obase + ot) * 16 + lm;
    float bv = bias[192 + col];
#pragma unroll
    for (int r = 0; r < 8; ++r) {
      float v = acc2[ot][r] + bv;
      float a = fabsf(v) - LAM_C;
      v = (a > 0.0f) ? copysignf(a, v) : 0.0f;   // softshrink
      stag[(mbase + r) * AROW + col] = v;
    }
  }
  __syncthreads();
  // coalesced float2 write-back (Re from col o, Im from col 96+o), in place
  for (int idx = tid; idx < 64 * BS; idx += 256) {
    int o = idx >> 6;
    int p = idx & 63;
    int P = p0 + p;
    int eta = P / KEPT;
    int w = P - eta * KEPT;
    spec[((size_t)(c0 + o) * H_DIM + eta) * WC + w] =
        mkf2(stag[p * AROW + o], stag[p * AROW + BS + o]);
  }
}

// ---------------- K5: Hermitian fill + inverse row FFT + residual ----------------
__global__ void __launch_bounds__(256)
afno_irowfft_kernel(const f2* __restrict__ spec, const float* __restrict__ x,
                    float* __restrict__ out) {
  extern __shared__ char sm[];
  f2* bufA = (f2*)sm;
  f2* bufB = bufA + W_DIM;
  const int row = blockIdx.x;
  const int tid = threadIdx.x;
  const f2* sp = spec + (size_t)row * WC;
  f2 a = (tid < KEPT) ? sp[tid] : mkf2(0.0f, 0.0f);
  bufA[tid] = a;
  int b2 = tid + 256;
  f2 bv = mkf2(0.0f, 0.0f);
  if (b2 >= W_DIM - (KEPT - 1)) {        // bins 384..511 = conj(spec[1..128])
    f2 t = sp[W_DIM - b2];
    bv = mkf2(t.x, -t.y);
  }
  bufA[b2] = bv;
  f2* res = stockham_fft<W_DIM>(bufA, bufB, 1, 1.0f, tid, 256);
  const float s = 0.04419417382415922f;  // 1/sqrt(512)
  const float* xr = x + (size_t)row * W_DIM;
  float* orow = out + (size_t)row * W_DIM;
  orow[tid]       = res[tid].x * s + xr[tid];
  orow[tid + 256] = res[tid + 256].x * s + xr[tid + 256];
}

// ---------------- host launch ----------------
extern "C" void kernel_launch(void* const* d_in, const int* in_sizes, int n_in,
                              void* d_out, int out_size, void* d_ws, size_t ws_size,
                              hipStream_t stream) {
  (void)in_sizes; (void)n_in; (void)out_size; (void)ws_size;
  const float* x  = (const float*)d_in[0];
  const float* w1 = (const float*)d_in[1];
  const float* b1 = (const float*)d_in[2];
  const float* w2 = (const float*)d_in[3];
  const float* b2 = (const float*)d_in[4];
  float* out = (float*)d_out;

  char* ws = (char*)d_ws;
  f2* spec = (f2*)ws;                                              // 768*256*257 f2
  const size_t spec_bytes = (size_t)C_DIM * H_DIM * WC * sizeof(f2);
  _Float16* wexp = (_Float16*)(ws + spec_bytes);                   // 8*2*192*192 f16
  const size_t wexp_bytes = (size_t)NB * 2 * 192 * 192 * sizeof(_Float16);
  float* bexp = (float*)(ws + spec_bytes + wexp_bytes);            // 8*2*192 f32

  hipFuncSetAttribute((const void*)afno_mlp_kernel,
                      hipFuncAttributeMaxDynamicSharedMemorySize, 131072);
  hipFuncSetAttribute((const void*)afno_colfft_kernel,
                      hipFuncAttributeMaxDynamicSharedMemorySize, 65536);

  afno_expand_kernel<<<dim3(NB, 2), 256, 0, stream>>>(w1, b1, w2, b2, wexp, bexp);
  afno_rowfft_kernel<<<C_DIM * H_DIM, 256, 2 * W_DIM * sizeof(f2), stream>>>(x, spec);
  afno_colfft_kernel<<<dim3(C_DIM, 9), 256, 2 * 16 * H_DIM * sizeof(f2), stream>>>(spec, -1.0f);
  afno_mlp_kernel<<<NB * 516, 256, MLP_LDS, stream>>>(spec, wexp, bexp);
  afno_colfft_kernel<<<dim3(C_DIM, 9), 256, 2 * 16 * H_DIM * sizeof(f2), stream>>>(spec, 1.0f);
  afno_irowfft_kernel<<<C_DIM * H_DIM, 256, 2 * W_DIM * sizeof(f2), stream>>>(spec, x, out);
}
